// Head_50989851738631
// MI455X (gfx1250) — matrix operand order repaired and checked by reading, not compiled, not run on hardware
//
#include <hip/hip_runtime.h>
#include <hip/hip_bf16.h>
#include <stdint.h>

#define B_  4
#define T_  4096
#define C_  1024
#define H_  64
#define KSTRIDE 72   // padded LDS row stride (elements): 144B = 36 dwords -> conflict-free b128

typedef __bf16 bf16;
typedef __attribute__((ext_vector_type(16))) __bf16 v16bf;
typedef __attribute__((ext_vector_type(8)))  __bf16 v8bf;
typedef __attribute__((ext_vector_type(8)))  float  v8f;
typedef __attribute__((ext_vector_type(4)))  float  v4f;

static __device__ __forceinline__ v16bf ld16(const bf16* p) {
    v8bf lo = *(const v8bf*)(p);
    v8bf hi = *(const v8bf*)(p + 8);
    return __builtin_shufflevector(lo, hi, 0,1,2,3,4,5,6,7,8,9,10,11,12,13,14,15);
}
static __device__ __forceinline__ v16bf ld16s(const bf16* p0, const bf16* p1) {
    v8bf lo = *(const v8bf*)(p0);
    v8bf hi = *(const v8bf*)(p1);
    return __builtin_shufflevector(lo, hi, 0,1,2,3,4,5,6,7,8,9,10,11,12,13,14,15);
}
static __device__ __forceinline__ v8f vzero8() {
    v8f z = {0.f,0.f,0.f,0.f,0.f,0.f,0.f,0.f};
    return z;
}
static __device__ __forceinline__ v8f bwmma(v16bf a, v16bf b, v8f c) {
    return __builtin_amdgcn_wmma_f32_16x16x32_bf16(false, a, false, b, (short)0, c, false, false);
}
static __device__ __forceinline__ float rmax16(float v) {
    v = fmaxf(v, __shfl_xor(v, 1));
    v = fmaxf(v, __shfl_xor(v, 2));
    v = fmaxf(v, __shfl_xor(v, 4));
    v = fmaxf(v, __shfl_xor(v, 8));
    return v;
}
// Async global->LDS copy, 16B per lane (ASYNCcnt-tracked). GVS mode: SGPR base + VGPR offset.
static __device__ __forceinline__ void async_b128(uint32_t lds_addr, uint32_t voff, uint64_t sbase) {
    asm volatile("global_load_async_to_lds_b128 %0, %1, %2"
                 :: "v"(lds_addr), "v"(voff), "s"(sbase) : "memory");
}
static __device__ __forceinline__ void wait_async0() {
    asm volatile("s_wait_asynccnt 0" ::: "memory");
}

// ---- Kernel 0: W [C,H] f32 -> WT [3][H,C] bf16 (Wq pre-scaled by C^-0.5) ----
__global__ void prep_w_kernel(const float* __restrict__ Wq, const float* __restrict__ Wk,
                              const float* __restrict__ Wv, bf16* __restrict__ wt) {
    int i = blockIdx.x * blockDim.x + threadIdx.x;
    if (i >= 3 * H_ * C_) return;
    int m = i >> 16;
    int rem = i & 65535;
    int h = rem >> 10;
    int c = rem & (C_ - 1);
    const float* W = (m == 0) ? Wq : ((m == 1) ? Wk : Wv);
    float v = W[c * H_ + h];
    if (m == 0) v *= 0.03125f;    // C^-0.5 = 1/32 folded into q
    wt[i] = (bf16)v;
}

// ---- Kernel 1: q,k,v projections. One wave = one 16-row tile of [BT,C]x[C,3*64] ----
__global__ void __launch_bounds__(128) proj_kernel(const float* __restrict__ x,
                                                   const bf16* __restrict__ wt,
                                                   bf16* __restrict__ q,
                                                   bf16* __restrict__ k,
                                                   bf16* __restrict__ vt) {
    const int lane = threadIdx.x & 31;
    const int wv   = threadIdx.x >> 5;
    const int tile = blockIdx.x * 4 + wv;        // 0..1023
    const int bt0  = tile * 16;
    const int n    = lane & 15;
    const int half = lane >> 4;
    const int koffA = half ? 8 : 0;
    const int koffB = half ? 8 : 0;

    v8f acc[3][4];
    #pragma unroll
    for (int m = 0; m < 3; ++m)
        #pragma unroll
        for (int t = 0; t < 4; ++t) acc[m][t] = vzero8();

    const float* xrow = x + (size_t)(bt0 + n) * C_;

    for (int kc = 0; kc < C_; kc += 32) {
        const float* px = xrow + kc + koffA;
        v4f f0 = *(const v4f*)(px);
        v4f f1 = *(const v4f*)(px + 4);
        v4f f2 = *(const v4f*)(px + 16);
        v4f f3 = *(const v4f*)(px + 20);
        v16bf a;
        #pragma unroll
        for (int j = 0; j < 4; ++j) {
            a[j]      = (__bf16)f0[j];
            a[j + 4]  = (__bf16)f1[j];
            a[j + 8]  = (__bf16)f2[j];
            a[j + 12] = (__bf16)f3[j];
        }
        #pragma unroll
        for (int m = 0; m < 3; ++m) {
            #pragma unroll
            for (int t = 0; t < 4; ++t) {
                const bf16* pw = wt + m * (H_ * C_) + (16 * t + n) * C_ + kc + koffB;
                v16bf b = ld16s(pw, pw + 16);
                acc[m][t] = bwmma(a, b, acc[m][t]);
            }
        }
    }

    #pragma unroll
    for (int t = 0; t < 4; ++t) {
        #pragma unroll
        for (int r = 0; r < 8; ++r) {
            int rowl = r + 8 * half;
            int gr   = bt0 + rowl;
            int col  = 16 * t + n;
            q[gr * H_ + col] = (bf16)acc[0][t][r];
            k[gr * H_ + col] = (bf16)acc[1][t][r];
            int bb = gr >> 12;
            int tt = gr & (T_ - 1);
            vt[(bb * H_ + col) * T_ + tt] = (bf16)acc[2][t][r];
        }
    }
}

// ---- Kernel 2: flash attention, 64-key blocks, async double-buffered K/V in LDS ----
// Workgroup = 4 waves = 4 consecutive q-tiles of the same batch; K/V blocks shared.
__global__ void __launch_bounds__(128) attn_kernel(const bf16* __restrict__ q,
                                                   const bf16* __restrict__ k,
                                                   const bf16* __restrict__ vt,
                                                   float* __restrict__ out) {
    // K double-buffer | V double-buffer | per-wave P tiles   (46080 B total)
    __shared__ __align__(16) __bf16 smem[4 * 64 * KSTRIDE + 4 * 16 * KSTRIDE];
    const int KOFF = 0;
    const int VOFF = 2 * 64 * KSTRIDE;
    const int POFF = 4 * 64 * KSTRIDE;

    const int tid  = threadIdx.x;
    const int lane = tid & 31;
    const int wv   = tid >> 5;
    const int tile = blockIdx.x * 4 + wv;        // 0..1023
    const int b    = tile >> 8;                  // 256 q-tiles per batch
    const int q0   = (tile & 255) * 16;
    const int n    = lane & 15;
    const int half = lane >> 4;
    const int koffA = half ? 8 : 0;              // A frag K split
    const int koffB = half ? 8 : 0;              // B frag K split

    const bf16* qp = q  + (size_t)b * T_ * H_;
    const bf16* kp = k  + (size_t)b * T_ * H_;
    const bf16* vp = vt + (size_t)b * H_ * T_;
    bf16* pl = smem + POFF + wv * (16 * KSTRIDE);

    // Cooperative async-copy chunk map: 512 chunks of 16B per 8KB block, 4 per thread.
    uint32_t voffK[4], voffV[4], ldsK[2][4], ldsV[2][4];
    #pragma unroll
    for (int r = 0; r < 4; ++r) {
        int j   = tid + 128 * r;
        int row = j >> 3;                        // key (for K) or head-dim (for V)
        int c8  = j & 7;                         // 8-element column chunk
        voffK[r] = (uint32_t)(row * (H_ * 2) + c8 * 16);
        voffV[r] = (uint32_t)(row * (T_ * 2) + c8 * 16);
        #pragma unroll
        for (int bu = 0; bu < 2; ++bu) {
            ldsK[bu][r] = (uint32_t)(uintptr_t)(smem + KOFF + bu * 64 * KSTRIDE + row * KSTRIDE + c8 * 8);
            ldsV[bu][r] = (uint32_t)(uintptr_t)(smem + VOFF + bu * 64 * KSTRIDE + row * KSTRIDE + c8 * 8);
        }
    }
    auto issue_kv = [&](int kb, int bu) {
        uint64_t baseK = (uint64_t)(uintptr_t)(kp + (size_t)kb * H_);
        uint64_t baseV = (uint64_t)(uintptr_t)(vp + kb);
        #pragma unroll
        for (int r = 0; r < 4; ++r) async_b128(ldsK[bu][r], voffK[r], baseK);
        #pragma unroll
        for (int r = 0; r < 4; ++r) async_b128(ldsV[bu][r], voffV[r], baseV);
    };

    // Q A-fragments, loaded once
    const bf16* pq = qp + (q0 + n) * H_ + koffA;
    v16bf qa0 = ld16s(pq,      pq + 16);
    v16bf qa1 = ld16s(pq + 32, pq + 48);

    v16bf ones;
    #pragma unroll
    for (int j = 0; j < 16; ++j) ones[j] = (__bf16)1.0f;

    float mr[8], lr[8];
    v8f O[4];
    #pragma unroll
    for (int r = 0; r < 8; ++r) { mr[r] = -1e30f; lr[r] = 0.f; }
    #pragma unroll
    for (int t = 0; t < 4; ++t) O[t] = vzero8();

    const int kend_own = q0 + 16;
    const int kend_wg  = (((blockIdx.x * 4 + 3) & 255) * 16) + 16;   // max over WG's waves

    issue_kv(0, 0);
    for (int kb = 0; kb < kend_wg; kb += 64) {
        const int bu = (kb >> 6) & 1;
        wait_async0();                 // my loads for buffer bu are in LDS
        __syncthreads();               // => everyone's loads for bu are in LDS
        if (kb + 64 < kend_wg) issue_kv(kb + 64, bu ^ 1);   // prefetch overlaps compute

        if (kb < kend_own) {
            const bf16* kbuf = smem + KOFF + bu * 64 * KSTRIDE;
            const bf16* vbuf = smem + VOFF + bu * 64 * KSTRIDE;

            // S = Q K^T for four 16-key subtiles
            v8f s[4];
            #pragma unroll
            for (int st = 0; st < 4; ++st) {
                const bf16* kr = kbuf + (16 * st + n) * KSTRIDE + koffB;
                v16bf b0 = ld16s(kr, kr + 16);
                v16bf b1 = ld16s(kr + 32, kr + 48);
                v8f acc = vzero8();
                acc = bwmma(qa0, b0, acc);
                acc = bwmma(qa1, b1, acc);
                s[st] = acc;
            }

            const bool needmask = (kb + 63 > q0);
            float alpha[8];
            #pragma unroll
            for (int r = 0; r < 8; ++r) {
                float sr[4];
                #pragma unroll
                for (int st = 0; st < 4; ++st) sr[st] = s[st][r];
                if (needmask) {
                    int qrow = q0 + r + 8 * half;
                    #pragma unroll
                    for (int st = 0; st < 4; ++st)
                        if (kb + 16 * st + n > qrow) sr[st] = -1e30f;
                }
                float rowm = fmaxf(fmaxf(sr[0], sr[1]), fmaxf(sr[2], sr[3]));
                rowm = rmax16(rowm);
                float mnew = fmaxf(mr[r], rowm);
                alpha[r] = __expf(mr[r] - mnew);
                mr[r] = mnew;
                int rowl = r + 8 * half;
                #pragma unroll
                for (int st = 0; st < 4; ++st)
                    pl[rowl * KSTRIDE + 16 * st + n] = (__bf16)__expf(sr[st] - mnew);
                #pragma unroll
                for (int t = 0; t < 4; ++t) O[t][r] *= alpha[r];
            }

            // P as A-fragments (keys 0..31 / 32..63); DS ops in-order within wave
            const bf16* pa = pl + n * KSTRIDE + koffA;
            v16bf A0 = ld16s(pa,      pa + 16);
            v16bf A1 = ld16s(pa + 32, pa + 48);

            // row sums on the matrix pipe: l_part = P * ones (no shuffles)
            v8f rs = vzero8();
            rs = bwmma(A0, ones, rs);
            rs = bwmma(A1, ones, rs);
            #pragma unroll
            for (int r = 0; r < 8; ++r) lr[r] = lr[r] * alpha[r] + rs[r];

            // O += P V
            #pragma unroll
            for (int t = 0; t < 4; ++t) {
                const bf16* pv = vbuf + (16 * t + n) * KSTRIDE + koffB;
                v16bf vb0 = ld16s(pv, pv + 16);
                v16bf vb1 = ld16s(pv + 32, pv + 48);
                O[t] = bwmma(A0, vb0, O[t]);
                O[t] = bwmma(A1, vb1, O[t]);
            }
        }
        __syncthreads();               // readers done before bu is refilled next round
    }

    #pragma unroll
    for (int t = 0; t < 4; ++t) {
        #pragma unroll
        for (int r = 0; r < 8; ++r) {
            int rowl = r + 8 * half;
            float inv = 1.0f / lr[r];
            out[((size_t)b * T_ + q0 + rowl) * H_ + 16 * t + n] = O[t][r] * inv;
        }
    }
}

extern "C" void kernel_launch(void* const* d_in, const int* in_sizes, int n_in,
                              void* d_out, int out_size, void* d_ws, size_t ws_size,
                              hipStream_t stream) {
    const float* x  = (const float*)d_in[0];
    const float* Wq = (const float*)d_in[1];
    const float* Wk = (const float*)d_in[2];
    const float* Wv = (const float*)d_in[3];
    float* out = (float*)d_out;

    // workspace (bf16 elements): WT[3][H,C] | q[BT,H] | k[BT,H] | vT[B,H,T]
    bf16* ws  = (bf16*)d_ws;
    bf16* wt  = ws;
    bf16* qb  = wt + (size_t)3 * H_ * C_;
    bf16* kb  = qb + (size_t)B_ * T_ * H_;
    bf16* vtb = kb + (size_t)B_ * T_ * H_;

    {
        int total = 3 * H_ * C_;
        prep_w_kernel<<<(total + 255) / 256, 256, 0, stream>>>(Wq, Wk, Wv, wt);
    }
    proj_kernel<<<256, 128, 0, stream>>>(x, wt, qb, kb, vtb);
    attn_kernel<<<256, 128, 0, stream>>>(qb, kb, vtb, out);
}
